// MLPDecoder_53008486367988
// MI455X (gfx1250) — matrix operand order repaired and checked by reading, not compiled, not run on hardware
//
#include <hip/hip_runtime.h>

// MI455X / gfx1250: wave32, WMMA 16x16x32 f16 -> f32 accumulate.
// Fused gather + [E,256]x[256,128] GEMM + ReLU + [128]-dot + sigmoid.
// Node table (51 MB) is L2-resident; W1 staged once per block into LDS as f16.

typedef __attribute__((ext_vector_type(16))) _Float16 v16h;
typedef __attribute__((ext_vector_type(8)))  _Float16 v8h;
typedef __attribute__((ext_vector_type(8)))  float    v8f;

#define NHID   128
#define KTOT   256          // 2*NHID (concat of x,y rows)
#define KP     264          // padded LDS row stride in halfs: 4-bank shift/row, 16B aligned
#define TILE_M 16

__launch_bounds__(256)
__global__ void mlp_decoder_wmma(const float* __restrict__ inputs,
                                 const int*   __restrict__ x_idx,
                                 const int*   __restrict__ y_idx,
                                 const float* __restrict__ W1,     // [256][128] row-major (k,n)
                                 const float* __restrict__ bias1,  // [128]
                                 const float* __restrict__ W2,     // [128]
                                 const float* __restrict__ bias2,  // [1]
                                 float* __restrict__ out,          // [E]
                                 int E, int ntiles)
{
    // W1 transposed to f16: w1t[n][k], k contiguous. 128*264*2 = 67584 B LDS.
    __shared__ _Float16 w1t[NHID * KP];

    const int tid = threadIdx.x;

    // ---- stage W1 (f32 global -> f16 LDS, transposed) once per block ----
    for (int i = tid; i < KTOT * NHID; i += 256) {
        int k = i >> 7;        // row of W1 (K index)
        int n = i & 127;       // col of W1 (N index)
        w1t[n * KP + k] = (_Float16)W1[i];
    }
    __syncthreads();

    const int lane = tid & 31;
    const int wave = tid >> 5;
    const int g    = lane >> 4;   // half-wave group (selects K sub-slice / M sub-rows)
    const int n16  = lane & 15;   // column within N-chunk == row slot within M-tile

    // per-lane bias1 / W2 for each of the 8 N-chunks
    float b1v[8], w2v[8];
    #pragma unroll
    for (int j = 0; j < 8; ++j) {
        b1v[j] = bias1[j * 16 + n16];
        w2v[j] = W2[j * 16 + n16];
    }
    const float b2 = bias2[0];

    // grid-stride over 16-edge tiles; one tile per wave per iteration
    for (int tile = blockIdx.x * 8 + wave; tile < ntiles; tile += gridDim.x * 8) {
        // lane's M-row slot gathers its edge's endpoint rows
        int e  = tile * TILE_M + n16;
        int eg = (e < E) ? e : (E - 1);          // clamp (stores are guarded)
        const float* xr = inputs + (long long)x_idx[eg] * NHID;
        const float* yr = inputs + (long long)y_idx[eg] * NHID;

        // ---- build 8 A fragments: 16x32 f16, K = 256 over concat(x,y) ----
        // ISA layout: lane group g holds K = kc*32 + g*8 + {0..7} in elems 0..7
        //             and        K = kc*32 + 16 + g*8 + {0..7} in elems 8..15
        v16h a[8];
        #pragma unroll
        for (int kc = 0; kc < 8; ++kc) {
            const float* row = (kc < 4) ? xr : yr;
            const int off = ((kc & 3) * 32) + g * 8;
            const float4 p0 = *(const float4*)(row + off);
            const float4 p1 = *(const float4*)(row + off + 4);
            const float4 p2 = *(const float4*)(row + off + 16);
            const float4 p3 = *(const float4*)(row + off + 20);
            a[kc][0]  = (_Float16)p0.x; a[kc][1]  = (_Float16)p0.y;
            a[kc][2]  = (_Float16)p0.z; a[kc][3]  = (_Float16)p0.w;
            a[kc][4]  = (_Float16)p1.x; a[kc][5]  = (_Float16)p1.y;
            a[kc][6]  = (_Float16)p1.z; a[kc][7]  = (_Float16)p1.w;
            a[kc][8]  = (_Float16)p2.x; a[kc][9]  = (_Float16)p2.y;
            a[kc][10] = (_Float16)p2.z; a[kc][11] = (_Float16)p2.w;
            a[kc][12] = (_Float16)p3.x; a[kc][13] = (_Float16)p3.y;
            a[kc][14] = (_Float16)p3.z; a[kc][15] = (_Float16)p3.w;
        }

        // ---- layer 1 (WMMA) fused with layer 2 partial dot ----
        // nc stays a REAL loop: keeps only 8 B fragments live at a time so
        // they stream from LDS (ds_load_b128) instead of spilling to scratch.
        float acc[8] = {0.f, 0.f, 0.f, 0.f, 0.f, 0.f, 0.f, 0.f};
        #pragma unroll 1
        for (int nc = 0; nc < 8; ++nc) {
            const _Float16* wrow = &w1t[(nc * 16 + n16) * KP + g * 8];
            v8f c = {};
            #pragma unroll
            for (int kc = 0; kc < 8; ++kc) {
                // B fragment 32x16: lane col n16, K = kc*32 + g*8 + e (e<8), kc*32 + 16 + g*8 + (e-8) (e>=8)
                v8h blo = *(const v8h*)(wrow + kc * 32);
                v8h bhi = *(const v8h*)(wrow + kc * 32 + 16);
                v16h b  = __builtin_shufflevector(blo, bhi,
                              0, 1, 2, 3, 4, 5, 6, 7, 8, 9, 10, 11, 12, 13, 14, 15);
                c = __builtin_amdgcn_wmma_f32_16x16x32_f16(
                        /*neg_a=*/false, a[kc], /*neg_b=*/false, b,
                        /*c_mod=*/(short)0, c, /*reuse_a=*/false, /*reuse_b=*/false);
            }
            const float b1x = b1v[nc];
            const float w2x = w2v[nc];
            #pragma unroll
            for (int v = 0; v < 8; ++v) {
                float h = c[v] + b1x;          // C[m = v + 8*g][n = nc*16 + n16]
                h = (h > 0.f) ? h : 0.f;       // ReLU
                acc[v] += h * w2x;             // layer-2 partial dot
            }
        }

        // ---- reduce layer-2 partials across the 16 columns ----
        #pragma unroll
        for (int v = 0; v < 8; ++v) {
            float s = acc[v];
            s += __shfl_xor(s, 1);
            s += __shfl_xor(s, 2);
            s += __shfl_xor(s, 4);
            s += __shfl_xor(s, 8);
            acc[v] = s;
        }

        // lanes 0 (rows 0-7) and 16 (rows 8-15) write sigmoid outputs
        if (n16 == 0) {
            #pragma unroll
            for (int v = 0; v < 8; ++v) {
                int eo = tile * TILE_M + v + 8 * g;
                if (eo < E) {
                    float z = acc[v] + b2;
                    out[eo] = 1.0f / (1.0f + __expf(-z));
                }
            }
        }
    }
}

extern "C" void kernel_launch(void* const* d_in, const int* in_sizes, int n_in,
                              void* d_out, int out_size, void* d_ws, size_t ws_size,
                              hipStream_t stream) {
    const float* inputs = (const float*)d_in[0];
    const int*   x_idx  = (const int*)d_in[1];
    const int*   y_idx  = (const int*)d_in[2];
    const float* W1     = (const float*)d_in[3];
    const float* bias1  = (const float*)d_in[4];
    const float* W2     = (const float*)d_in[5];
    const float* bias2  = (const float*)d_in[6];
    float*       out    = (float*)d_out;

    const int E      = in_sizes[1];           // number of edges
    const int ntiles = (E + TILE_M - 1) / TILE_M;
    int blocks = (ntiles + 7) / 8;            // 8 waves (tiles) per 256-thread block
    if (blocks > 1024) blocks = 1024;         // persistent grid-stride blocks
    if (blocks < 1) blocks = 1;

    hipLaunchKernelGGL(mlp_decoder_wmma, dim3(blocks), dim3(256), 0, stream,
                       inputs, x_idx, y_idx, W1, bias1, W2, bias2, out, E, ntiles);
}
